// DecGCN_88888643158468
// MI455X (gfx1250) — matrix operand order repaired and checked by reading, not compiled, not run on hardware
//
#include <hip/hip_runtime.h>
#include <hip/hip_bf16.h>

// ---------------- problem constants (match reference) ----------------
#define PN0   16000
#define PN1   12000
#define PN2   6000
#define PK0   10
#define PK1   5
#define PH    256
#define POUT  128
#define PEIN  128

typedef __attribute__((ext_vector_type(16))) _Float16 v16h;
typedef __attribute__((ext_vector_type(8)))  _Float16 v8h;
typedef __attribute__((ext_vector_type(8)))  float    v8f;

// =====================================================================
// B pre-swizzle: f32 B[K,N] -> f16 fragment-ordered panels.
// Element t: i = t&15 (half idx), lane = (t>>4)&31, then (ks, nt).
// Lane layout used here (contiguous k, not ISA table order): lane = column n0+(lane&15); halves i are
// consecutive K with +16 offset for lanes 16-31 ((lane&16)).
// Each lane's 16-half fragment is contiguous (32B) in Bsw.
// =====================================================================
__global__ __launch_bounds__(256)
void decgcn_swizzle_b(const float* __restrict__ B, _Float16* __restrict__ Bsw,
                      int K, int N)
{
    int t = blockIdx.x * blockDim.x + threadIdx.x;
    if (t >= K * N) return;
    int i    = t & 15;
    int lane = (t >> 4) & 31;
    int rest = t >> 9;
    int ksteps = K >> 5;
    int ks = rest % ksteps;
    int nt = rest / ksteps;
    int k = (ks << 5) + (lane & 16) + i;
    int n = (nt << 4) + (lane & 15);
    Bsw[t] = (_Float16)B[(size_t)k * N + n];
}

// =====================================================================
// f16 WMMA GEMM:  C[M,N] = A[M,KK] * B[KK,N] (+bias[N])
// A row-major f16 (lda == KK); B pre-swizzled fragment panels.
// One wave per 16x16 C tile, 8 waves stacked along M per block.
// Requires M % 16 == 0 (true for all call sites: 16000, 6000).
// Fully unrolled K loop -> KK/32 straight-line v_wmma per wave.
// =====================================================================
template<int KK>
__global__ __launch_bounds__(256)
void decgcn_gemm_wmma(const _Float16* __restrict__ A,
                      const _Float16* __restrict__ Bsw,
                      const float* __restrict__ bias,
                      float* __restrict__ C, _Float16* __restrict__ C16,
                      int ldc, int M)
{
    constexpr int KSTEPS = KK / 32;
    const int lane = threadIdx.x & 31;
    const int wave = threadIdx.x >> 5;
    const int m0 = blockIdx.x * 128 + wave * 16;
    const int nt = blockIdx.y;
    if (m0 >= M) return;                        // wave-uniform (grid tail)

    const int hs8 = (lane & 16) >> 1;           // 0 (lanes 0-15) or 8
    const int r   = m0 + (lane & 15);
    const int col = nt * 16 + (lane & 15);

    // A fragment base: halves [0..15]=K+0..15 (lo lanes),
    //                  [0..15]=K+16..31 (hi lanes)
    const _Float16* ap = A + (size_t)r * KK + (lane & 16);
    // B fragment base: contiguous 16 halves per lane per k-step
    const _Float16* bbase = Bsw + (size_t)nt * KSTEPS * 512 + (size_t)lane * 16;

    v8f acc = {};
#pragma unroll
    for (int ks = 0; ks < KSTEPS; ++ks) {
        v8h lo = *(const v8h*)(ap + ks * 32);
        v8h hi = *(const v8h*)(ap + ks * 32 + 8);
        v16h a;
#pragma unroll
        for (int i = 0; i < 8; ++i) { a[i] = lo[i]; a[8 + i] = hi[i]; }
        v16h b = *(const v16h*)(bbase + ks * 512);
        acc = __builtin_amdgcn_wmma_f32_16x16x32_f16(
            false, a, false, b, (short)0, acc, false, false);
    }

    const float bv = bias ? bias[col] : 0.0f;
    const int rbase = m0 + ((lane & 16) >> 1);  // C/D: VGPR v -> M=v (+8 hi half)
    float*    cp  = C + (size_t)rbase * ldc + col;
    _Float16* cp16 = C16 ? (C16 + (size_t)rbase * ldc + col) : nullptr;
#pragma unroll
    for (int v = 0; v < 8; ++v) {
        float o = acc[v] + bv;
        cp[(size_t)v * ldc] = o;
        if (cp16) cp16[(size_t)v * ldc] = (_Float16)o;
    }
}

// =====================================================================
// Embedding gather + f16 convert:  E[n,128] = [e0[x0] | e1[x1] | e2[x2]]
// =====================================================================
__global__ __launch_bounds__(256)
void decgcn_gather_embed(const int* __restrict__ x,
                         const float* __restrict__ e0,
                         const float* __restrict__ e1,
                         const float* __restrict__ e2,
                         _Float16* __restrict__ E, int n)
{
    int t = blockIdx.x * blockDim.x + threadIdx.x;
    if (t >= n * PEIN) return;
    int row = t >> 7, c = t & 127;
    float v;
    if (c < 64)      v = e0[(size_t)x[row * 3 + 0] * 64 + c];
    else if (c < 96) v = e1[(size_t)x[row * 3 + 1] * 32 + (c - 64)];
    else             v = e2[(size_t)x[row * 3 + 2] * 32 + (c - 96)];
    E[t] = (_Float16)v;
}

// =====================================================================
// Co-attention aggregation, algebraically reduced.
// Per dst node i (one block, 256 threads = H channels):
//   D,Q  = gathered K x 256 rows;  L = D Q^T (K x K)
//   AC = softmax(L, rows), AS = softmax(L^T, rows)
//   s = 1^T AS ; u = AC^T s
//   colsum full = [ 1^T Q | s^T D | u^T Q ]  (768 values)
//   neigh[j] = (F[3j]+F[3j+1]+F[3j+2]) / (3K)
//   out[i] = feat[i] + neigh  (optional relu)
// =====================================================================
__global__ __launch_bounds__(256)
void decgcn_coagg(const float* __restrict__ feat,
                  const int* __restrict__ idx_sim,
                  const int* __restrict__ idx_cor,
                  float* __restrict__ out, _Float16* __restrict__ out16,
                  int K, int mode_is_cor, int do_relu)
{
    __shared__ float Dsh[16][PH];     // row-major D
    __shared__ float QT[PH][16];      // transposed Q (bank-friendly both loops)
    __shared__ float Lm[16][16];
    __shared__ float ACm[16][16];
    __shared__ float ASm[16][16];
    __shared__ float sv[16], uv[16];
    __shared__ float F[3 * PH];

    const int i = blockIdx.x;
    const int t = threadIdx.x;        // channel 0..255

    for (int k = 0; k < K; ++k) {
        int is = idx_sim[(size_t)i * K + k];
        int ic = idx_cor[(size_t)i * K + k];
        int di = mode_is_cor ? is : ic;
        int qi = mode_is_cor ? ic : is;
        Dsh[k][t] = feat[(size_t)di * PH + t];
        QT[t][k]  = feat[(size_t)qi * PH + t];
    }
    __syncthreads();

    if (t < K * K) {                  // K^2 <= 100 dot products, len 256
        int k = t / K, m = t % K;
        float acc = 0.f;
        for (int c = 0; c < PH; ++c) acc += Dsh[k][c] * QT[c][m];
        Lm[k][m] = acc;
    }
    __syncthreads();

    if (t < K) {                      // softmax row t of L -> AC; of L^T -> AS
        float mx = -3.0e38f, sm = 0.f;
        for (int m = 0; m < K; ++m) mx = fmaxf(mx, Lm[t][m]);
        for (int m = 0; m < K; ++m) { float e = __expf(Lm[t][m] - mx); ACm[t][m] = e; sm += e; }
        float inv = 1.f / sm;
        for (int m = 0; m < K; ++m) ACm[t][m] *= inv;

        mx = -3.0e38f; sm = 0.f;
        for (int m = 0; m < K; ++m) mx = fmaxf(mx, Lm[m][t]);
        for (int m = 0; m < K; ++m) { float e = __expf(Lm[m][t] - mx); ASm[t][m] = e; sm += e; }
        inv = 1.f / sm;
        for (int m = 0; m < K; ++m) ASm[t][m] *= inv;
    }
    __syncthreads();

    if (t < K) { float a = 0.f; for (int k = 0; k < K; ++k) a += ASm[k][t]; sv[t] = a; }
    __syncthreads();
    if (t < K) { float a = 0.f; for (int k = 0; k < K; ++k) a += ACm[k][t] * sv[k]; uv[t] = a; }
    __syncthreads();

    {
        float qs = 0.f, ds = 0.f, cqs = 0.f;
        for (int k = 0; k < K; ++k) {
            float qv = QT[t][k];
            qs  += qv;
            cqs += uv[k] * qv;
            ds  += sv[k] * Dsh[k][t];
        }
        F[t] = qs; F[PH + t] = ds; F[2 * PH + t] = cqs;
    }
    __syncthreads();

    float nv = (F[3 * t] + F[3 * t + 1] + F[3 * t + 2]) / (3.0f * (float)K);
    float o  = feat[(size_t)i * PH + t] + nv;
    if (do_relu) o = fmaxf(o, 0.f);
    out[(size_t)i * PH + t] = o;
    if (out16) out16[(size_t)i * PH + t] = (_Float16)o;
}

// =====================================================================
// Integration elementwise kernels
// =====================================================================
__global__ __launch_bounds__(256)
void decgcn_make_z1(const float* __restrict__ sim, const float* __restrict__ cor,
                    const float* __restrict__ P, const float* __restrict__ Qm,
                    const float* __restrict__ a1p,
                    _Float16* __restrict__ z1s, _Float16* __restrict__ z1c, int n)
{
    int t = blockIdx.x * blockDim.x + threadIdx.x;
    if (t >= n) return;
    float a1 = a1p[0], w = 1.f - a1;
    z1s[t] = (_Float16)(w * sim[t] + a1 * P[t]);
    z1c[t] = (_Float16)(w * cor[t] + a1 * Qm[t]);
}

__global__ __launch_bounds__(256)
void decgcn_integrate(const float* __restrict__ sim, const float* __restrict__ cor,
                      const float* __restrict__ P, const float* __restrict__ Qm,
                      const float* __restrict__ R1, const float* __restrict__ R2,
                      const float* __restrict__ a2p, const float* __restrict__ b2p,
                      float* __restrict__ out, int n)
{
    int t = blockIdx.x * blockDim.x + threadIdx.x;
    if (t >= n) return;
    float a2 = a2p[0], b2 = b2p[0], w = 1.f - a2 - b2;
    out[t]     = w * sim[t] + a2 * P[t]  + b2 * R1[t];   // z2sim
    out[n + t] = w * cor[t] + a2 * Qm[t] + b2 * R2[t];   // z2cor
}

// =====================================================================
// Host orchestration
// =====================================================================
extern "C" void kernel_launch(void* const* d_in, const int* in_sizes, int n_in,
                              void* d_out, int out_size, void* d_ws, size_t ws_size,
                              hipStream_t stream)
{
    const int*   x    = (const int*)d_in[0];
    const int*   is0  = (const int*)d_in[1];
    const int*   ic0  = (const int*)d_in[2];
    const int*   is1  = (const int*)d_in[3];
    const int*   ic1  = (const int*)d_in[4];
    const float* emb0[2] = { (const float*)d_in[5],  (const float*)d_in[12] };
    const float* emb1[2] = { (const float*)d_in[6],  (const float*)d_in[13] };
    const float* emb2[2] = { (const float*)d_in[7],  (const float*)d_in[14] };
    const float* Win [2] = { (const float*)d_in[8],  (const float*)d_in[15] };
    const float* binv[2] = { (const float*)d_in[9],  (const float*)d_in[16] };
    const float* Wout[2] = { (const float*)d_in[10], (const float*)d_in[17] };
    const float* bout[2] = { (const float*)d_in[11], (const float*)d_in[18] };
    const float* Ws2c = (const float*)d_in[19];
    const float* Wc2s = (const float*)d_in[20];
    const float* a1p  = (const float*)d_in[21];
    const float* a2p  = (const float*)d_in[22];
    const float* b2p  = (const float*)d_in[23];

    // ---- workspace carve (bump allocator, 256B aligned) ----
    char* wp = (char*)d_ws;
    auto carve = [&](size_t bytes) -> void* {
        void* r = (void*)wp;
        wp += (bytes + 255) & ~(size_t)255;
        return r;
    };
    _Float16* E16   = (_Float16*)carve((size_t)PN0 * PEIN * 2);
    float*    H0    = (float*)   carve((size_t)PN0 * PH * 4);   // FC1 out; reused for H2
    float*    H1    = (float*)   carve((size_t)PN1 * PH * 4);
    _Float16* H2h   = (_Float16*)carve((size_t)PN2 * PH * 2);
    _Float16* Winh0 = (_Float16*)carve((size_t)PEIN * PH * 2);
    _Float16* Winh1 = (_Float16*)carve((size_t)PEIN * PH * 2);
    _Float16* Wouth0= (_Float16*)carve((size_t)PH * POUT * 2);
    _Float16* Wouth1= (_Float16*)carve((size_t)PH * POUT * 2);
    _Float16* Ws2ch = (_Float16*)carve((size_t)POUT * POUT * 2);
    _Float16* Wc2sh = (_Float16*)carve((size_t)POUT * POUT * 2);
    float*    simF  = (float*)   carve((size_t)PN2 * POUT * 4);
    float*    corF  = (float*)   carve((size_t)PN2 * POUT * 4);
    _Float16* simH  = (_Float16*)carve((size_t)PN2 * POUT * 2);
    _Float16* corH  = (_Float16*)carve((size_t)PN2 * POUT * 2);
    float*    Pb    = (float*)   carve((size_t)PN2 * POUT * 4);
    float*    Qb    = (float*)   carve((size_t)PN2 * POUT * 4);
    float*    R1b   = (float*)   carve((size_t)PN2 * POUT * 4);
    float*    R2b   = (float*)   carve((size_t)PN2 * POUT * 4);
    _Float16* z1sH  = (_Float16*)carve((size_t)PN2 * POUT * 2);
    _Float16* z1cH  = (_Float16*)carve((size_t)PN2 * POUT * 2);
    float*    H2f   = H0;                                       // alias (H0 dead by then)

    _Float16* Winh[2]  = { Winh0, Winh1 };
    _Float16* Wouth[2] = { Wouth0, Wouth1 };

    // ---- weight conversions: f32 -> pre-swizzled f16 fragment panels ----
    auto swz = [&](const float* s, _Float16* d, int K, int N) {
        decgcn_swizzle_b<<<(K * N + 255) / 256, 256, 0, stream>>>(s, d, K, N);
    };
    swz(Win[0],  Winh[0],  PEIN, PH);
    swz(Win[1],  Winh[1],  PEIN, PH);
    swz(Wout[0], Wouth[0], PH, POUT);
    swz(Wout[1], Wouth[1], PH, POUT);
    swz(Ws2c, Ws2ch, POUT, POUT);
    swz(Wc2s, Wc2sh, POUT, POUT);

    // ---- per-branch pipeline (b=0 sim, b=1 cor) ----
    for (int b = 0; b < 2; ++b) {
        const int mode_is_cor = b;                 // 'sim' -> 0, 'cor' -> 1

        decgcn_gather_embed<<<(PN0 * PEIN + 255) / 256, 256, 0, stream>>>(
            x, emb0[b], emb1[b], emb2[b], E16, PN0);

        // H0 = E @ Win + bin       [16000 x 256], K=128
        decgcn_gemm_wmma<PEIN><<<dim3((PN0 + 127) / 128, PH / 16), 256, 0, stream>>>(
            E16, Winh[b], binv[b], H0, nullptr, PH, PN0);

        // level 0: relu(coagg(H0)) -> H1  [12000 x 256]
        decgcn_coagg<<<PN1, 256, 0, stream>>>(
            H0, is0, ic0, H1, nullptr, PK0, mode_is_cor, 1);

        // level 1: coagg(H1) -> H2 (f32 alias of H0) + f16   [6000 x 256]
        decgcn_coagg<<<PN2, 256, 0, stream>>>(
            H1, is1, ic1, H2f, H2h, PK1, mode_is_cor, 0);

        // branch out = H2 @ Wout + bout   [6000 x 128], K=256
        decgcn_gemm_wmma<PH><<<dim3((PN2 + 127) / 128, POUT / 16), 256, 0, stream>>>(
            H2h, Wouth[b], bout[b],
            b ? corF : simF, b ? corH : simH, POUT, PN2);
    }

    // ---- semantic integration ----
    const int nel = PN2 * POUT;
    dim3 gI((PN2 + 127) / 128, POUT / 16);

    // P = cor @ Wc2s ; Q = sim @ Ws2c
    decgcn_gemm_wmma<POUT><<<gI, 256, 0, stream>>>(corH, Wc2sh, nullptr,
                                                   Pb, nullptr, POUT, PN2);
    decgcn_gemm_wmma<POUT><<<gI, 256, 0, stream>>>(simH, Ws2ch, nullptr,
                                                   Qb, nullptr, POUT, PN2);

    decgcn_make_z1<<<(nel + 255) / 256, 256, 0, stream>>>(
        simF, corF, Pb, Qb, a1p, z1sH, z1cH, nel);

    // R1 = z1cor @ Wc2s ; R2 = z1sim @ Ws2c
    decgcn_gemm_wmma<POUT><<<gI, 256, 0, stream>>>(z1cH, Wc2sh, nullptr,
                                                   R1b, nullptr, POUT, PN2);
    decgcn_gemm_wmma<POUT><<<gI, 256, 0, stream>>>(z1sH, Ws2ch, nullptr,
                                                   R2b, nullptr, POUT, PN2);

    decgcn_integrate<<<(nel + 255) / 256, 256, 0, stream>>>(
        simF, corF, Pb, Qb, R1b, R2b, a2p, b2p, (float*)d_out, nel);
}
